// FullSelfAttention_21612275434296
// MI455X (gfx1250) — matrix operand order repaired and checked by reading, not compiled, not run on hardware
//
#include <hip/hip_runtime.h>
#include <hip/hip_bf16.h>

typedef __attribute__((ext_vector_type(16))) _Float16 v16h;
typedef __attribute__((ext_vector_type(8)))  float    v8f;

#define WMMA_F32_F16(a, b, c) \
  __builtin_amdgcn_wmma_f32_16x16x32_f16(false, (a), false, (b), (short)0, (c), false, false)

// ---------------- WMMA fragment builders (wave32, 16x16x32 f16) ----------------
// A matrix 16x32: lane holds row m=lane&15; VGPR q holds K = (q&3)*2 + (q>>2)*16 + half*8
__device__ __forceinline__ v16h frag_a(const _Float16* p, int strideH) {
  const int lane = threadIdx.x & 31;
  const int m = lane & 15, hf = lane >> 4;
  union { v16h v; unsigned u[8]; } r;
  const _Float16* base = p + m * strideH + hf * 8;
#pragma unroll
  for (int q = 0; q < 8; ++q) {
    int k = ((q & 3) * 2) + ((q >> 2) * 16);
    r.u[q] = *(const unsigned*)(base + k);
  }
  return r.v;
}

// B matrix 32x16 from LDS laid out [n][k] (k contiguous): lane holds col n=lane&15;
// VGPR q holds K = (q&3)*2 + (q>>2)*16 + half*8 (same as A)
__device__ __forceinline__ v16h frag_b(const _Float16* p, int strideH) {
  const int lane = threadIdx.x & 31;
  const int n = lane & 15, hf = lane >> 4;
  union { v16h v; unsigned u[8]; } r;
  const _Float16* base = p + n * strideH + hf * 8;
#pragma unroll
  for (int q = 0; q < 8; ++q)
    r.u[q] = *(const unsigned*)(base + ((q & 3) * 2) + ((q >> 2) * 16));
  return r.v;
}

#define BM 128
#define BN 64
#define TK 32
#define LDA 40   // TK + 8 pad halves (row = 80B: 8B/16B aligned sub-chunks stay aligned)

// ---------------- Kernel 1: QKV projection ----------------
// Y = x @ W_qkv + b_qkv ; scatter into Q/K/V f16 [B,H,T,D]
__global__ __launch_bounds__(256) void qkv_gemm(
    const float* __restrict__ x, const float* __restrict__ Wqkv,
    const float* __restrict__ bqkv,
    _Float16* __restrict__ Qh, _Float16* __restrict__ Kh, _Float16* __restrict__ Vh)
{
  __shared__ _Float16 As[BM * LDA];
  __shared__ _Float16 Bs[BN * LDA];
  const int tid = threadIdx.x;
  const int wave = tid >> 5;
  const int row0 = blockIdx.y * BM;
  const int col0 = blockIdx.x * BN;
  const int wr = (wave & 3) * 32;   // 4 row groups of 32
  const int wc = (wave >> 2) * 32;  // 2 col groups of 32
  v8f acc00 = {}, acc01 = {}, acc10 = {}, acc11 = {};
  for (int kt = 0; kt < 1024; kt += TK) {
    // A tile: 128x32 f32 -> f16, b128 global loads (4 floats/lane/iter)
#pragma unroll
    for (int i = 0; i < 4; ++i) {
      int e = i * 256 + tid;            // float4 chunk id, 1024 total
      int r = e >> 3, c4 = (e & 7) * 4;
      float4 f = *(const float4*)&x[(size_t)(row0 + r) * 1024 + kt + c4];
      union { _Float16 h[4]; unsigned long long u; } p;
      p.h[0] = (_Float16)f.x; p.h[1] = (_Float16)f.y;
      p.h[2] = (_Float16)f.z; p.h[3] = (_Float16)f.w;
      *(unsigned long long*)&As[r * LDA + c4] = p.u;
    }
    // B tile: 32x64 f32 -> f16 transposed to [n][k], b128 global loads
#pragma unroll
    for (int i = 0; i < 2; ++i) {
      int e = i * 256 + tid;            // float4 chunk id, 512 total
      int k = e >> 4, n4 = (e & 15) * 4;
      float4 f = *(const float4*)&Wqkv[(size_t)(kt + k) * 3072 + col0 + n4];
      Bs[(n4 + 0) * LDA + k] = (_Float16)f.x;
      Bs[(n4 + 1) * LDA + k] = (_Float16)f.y;
      Bs[(n4 + 2) * LDA + k] = (_Float16)f.z;
      Bs[(n4 + 3) * LDA + k] = (_Float16)f.w;
    }
    if (kt + TK < 1024) {  // prefetch next K tiles (global_prefetch_b8)
      __builtin_prefetch(&x[(size_t)(row0 + (tid >> 1)) * 1024 + kt + TK + (tid & 1) * 16], 0, 1);
      __builtin_prefetch(&Wqkv[(size_t)(kt + TK + (tid >> 6)) * 3072 + col0 + (tid & 63)], 0, 1);
    }
    __syncthreads();
    v16h a0 = frag_a(&As[wr * LDA], LDA);
    v16h a1 = frag_a(&As[(wr + 16) * LDA], LDA);
    v16h b0 = frag_b(&Bs[wc * LDA], LDA);
    v16h b1 = frag_b(&Bs[(wc + 16) * LDA], LDA);
    acc00 = WMMA_F32_F16(a0, b0, acc00);
    acc01 = WMMA_F32_F16(a0, b1, acc01);
    acc10 = WMMA_F32_F16(a1, b0, acc10);
    acc11 = WMMA_F32_F16(a1, b1, acc11);
    __syncthreads();
  }
  const int lane = tid & 31;
  const int n = lane & 15, hf = lane >> 4;
  const int bb = row0 >> 11;                       // block never crosses batch boundary
  const int t0 = (row0 & 2047) + wr + hf * 8;
#pragma unroll
  for (int ct = 0; ct < 2; ++ct) {
    int col = col0 + wc + ct * 16 + n;
    float bias = bqkv[col];
    int which = col >> 10;
    int cc = col & 1023;
    int hh = cc >> 6, d = cc & 63;
    _Float16* dst = (which == 0) ? Qh : (which == 1) ? Kh : Vh;
#pragma unroll
    for (int rt = 0; rt < 2; ++rt) {
      size_t dbase = ((size_t)(bb * 16 + hh) * 2048 + t0 + rt * 16) * 64 + d;
#pragma unroll
      for (int r = 0; r < 8; ++r) {
        float v = ((ct == 0) ? (rt == 0 ? acc00[r] : acc10[r])
                             : (rt == 0 ? acc01[r] : acc11[r])) + bias;
        dst[dbase + (size_t)r * 64] = (_Float16)v;
      }
    }
  }
}

// ---------------- Kernel 2: flash attention ----------------
#define KSTR 72  // 64 + 8 pad halves

__global__ __launch_bounds__(256) void attention(
    const _Float16* __restrict__ Qh, const _Float16* __restrict__ Kh,
    const _Float16* __restrict__ Vh, _Float16* __restrict__ AO)
{
  __shared__ _Float16 Ks[64 * KSTR];      // [key][d]
  __shared__ _Float16 Vt[64 * KSTR];      // [d][key] (transposed)
  __shared__ _Float16 Ps[8 * 16 * KSTR];  // per-wave P tile [m][key]
  const int tid = threadIdx.x;
  const int wave = tid >> 5, lane = tid & 31;
  const int hf = lane >> 4, ln = lane & 15;
  const int bh = blockIdx.y;
  const size_t base = (size_t)bh * 2048 * 64;
  const _Float16* Qp = Qh + base;
  const _Float16* Kp = Kh + base;
  const _Float16* Vp = Vh + base;
  const int qr = blockIdx.x * 128 + wave * 16;
  const v16h qa0 = frag_a(Qp + (size_t)qr * 64, 64);       // d 0..31
  const v16h qa1 = frag_a(Qp + (size_t)qr * 64 + 32, 64);  // d 32..63
  _Float16* PsW = &Ps[wave * 16 * KSTR];
  float m_i[8], l_i[8], alpha[8];
#pragma unroll
  for (int r = 0; r < 8; ++r) { m_i[r] = -3.0e38f; l_i[r] = 0.f; }
  v8f o0 = {}, o1 = {}, o2 = {}, o3 = {};
  const float scale = 0.125f;  // D^-0.5

  const unsigned ks_base = (unsigned)(size_t)(&Ks[0]);

  for (int kb = 0; kb < 2048; kb += 64) {
    // stage K chunk via CDNA5 async global->LDS (8192B = 512 x b128, 2 per lane)
    {
      const unsigned long long gK = (unsigned long long)(const void*)(Kp + (size_t)kb * 64);
#pragma unroll
      for (int i = 0; i < 2; ++i) {
        int e = i * 256 + tid;             // b128 chunk id 0..511
        int key = e >> 3, c = e & 7;       // 8 chunks per 64-half row
        unsigned lds_off = ks_base + (unsigned)(key * (KSTR * 2) + c * 16);
        unsigned goff = (unsigned)(key * 128 + c * 16);
        asm volatile("global_load_async_to_lds_b128 %0, %1, %2"
                     :: "v"(lds_off), "v"(goff), "s"(gK) : "memory");
      }
    }
    // stage V transposed: b128 global loads (merged from 4 dwords), b16 LDS scatter
    {
#pragma unroll
      for (int i = 0; i < 2; ++i) {
        int e = i * 256 + tid;             // 16B chunk id 0..511
        int key = e >> 3, d8 = (e & 7) * 8;
        const unsigned* s4 = (const unsigned*)(Vp + (size_t)(kb + key) * 64 + d8);
#pragma unroll
        for (int j = 0; j < 4; ++j) {
          union { unsigned u; _Float16 h[2]; } cv; cv.u = s4[j];
          Vt[(d8 + 2 * j)     * KSTR + key] = cv.h[0];
          Vt[(d8 + 2 * j + 1) * KSTR + key] = cv.h[1];
        }
      }
    }
    asm volatile("s_wait_asynccnt 0" ::: "memory");
    __syncthreads();

    // S = Q . K^T for 64 keys: 4 tiles of 16x16, each = 2 WMMAs over d
    v8f z = {};
    v8f s[4];
#pragma unroll
    for (int t = 0; t < 4; ++t) {
      v16h k0 = frag_b(&Ks[(t * 16) * KSTR], KSTR);       // d 0..31
      v16h k1 = frag_b(&Ks[(t * 16) * KSTR + 32], KSTR);  // d 32..63
      s[t] = WMMA_F32_F16(qa0, k0, z);
      s[t] = WMMA_F32_F16(qa1, k1, s[t]);
    }

    // online softmax: row max / sum across 4 tiles + 16-lane half-wave
#pragma unroll
    for (int r = 0; r < 8; ++r) {
      float v = fmaxf(fmaxf(s[0][r], s[1][r]), fmaxf(s[2][r], s[3][r]));
#pragma unroll
      for (int off = 1; off < 16; off <<= 1)
        v = fmaxf(v, __shfl_xor(v, off, 32));
      float mn = fmaxf(m_i[r], v * scale);
      alpha[r] = __expf(m_i[r] - mn);
      m_i[r] = mn;
      l_i[r] *= alpha[r];
    }
    float rsum[8];
#pragma unroll
    for (int r = 0; r < 8; ++r) rsum[r] = 0.f;
#pragma unroll
    for (int t = 0; t < 4; ++t)
#pragma unroll
      for (int r = 0; r < 8; ++r) {
        float ex = __expf(s[t][r] * scale - m_i[r]);
        rsum[r] += ex;
        PsW[(hf * 8 + r) * KSTR + t * 16 + ln] = (_Float16)ex;  // C-layout -> [m][key]
      }
#pragma unroll
    for (int r = 0; r < 8; ++r) {
      float v = rsum[r];
#pragma unroll
      for (int off = 1; off < 16; off <<= 1)
        v += __shfl_xor(v, off, 32);
      l_i[r] += v;
      o0[r] *= alpha[r]; o1[r] *= alpha[r]; o2[r] *= alpha[r]; o3[r] *= alpha[r];
    }

    // O += P(16x64) . V(64x64): per-wave LDS ops are in-order, no barrier needed
#pragma unroll
    for (int kk = 0; kk < 2; ++kk) {
      v16h pa = frag_a(PsW + kk * 32, KSTR);
      o0 = WMMA_F32_F16(pa, frag_b(&Vt[ 0 * KSTR + kk * 32], KSTR), o0);
      o1 = WMMA_F32_F16(pa, frag_b(&Vt[16 * KSTR + kk * 32], KSTR), o1);
      o2 = WMMA_F32_F16(pa, frag_b(&Vt[32 * KSTR + kk * 32], KSTR), o2);
      o3 = WMMA_F32_F16(pa, frag_b(&Vt[48 * KSTR + kk * 32], KSTR), o3);
    }
    __syncthreads();
  }

  const int bb = bh >> 4, hh = bh & 15;
#pragma unroll
  for (int r = 0; r < 8; ++r) {
    float inv = 1.0f / l_i[r];
    int row = qr + hf * 8 + r;
    size_t rb = ((size_t)(bb * 2048 + row)) * 1024 + hh * 64;
    AO[rb +  0 + ln] = (_Float16)(o0[r] * inv);
    AO[rb + 16 + ln] = (_Float16)(o1[r] * inv);
    AO[rb + 32 + ln] = (_Float16)(o2[r] * inv);
    AO[rb + 48 + ln] = (_Float16)(o3[r] * inv);
  }
}

// ---------------- Kernel 3: output projection ----------------
__global__ __launch_bounds__(256) void out_gemm(
    const _Float16* __restrict__ AO, const float* __restrict__ Wout,
    const float* __restrict__ bout, float* __restrict__ out)
{
  __shared__ _Float16 As[BM * LDA];
  __shared__ _Float16 Bs[BN * LDA];
  const int tid = threadIdx.x;
  const int wave = tid >> 5;
  const int row0 = blockIdx.y * BM;
  const int col0 = blockIdx.x * BN;
  const int wr = (wave & 3) * 32;
  const int wc = (wave >> 2) * 32;
  v8f acc00 = {}, acc01 = {}, acc10 = {}, acc11 = {};
  for (int kt = 0; kt < 1024; kt += TK) {
    // A tile: 128x32 f16, straight b128 copy global->LDS
#pragma unroll
    for (int i = 0; i < 2; ++i) {
      int e = i * 256 + tid;            // 16B chunk id, 512 total
      int r = e >> 2, c8 = (e & 3) * 8;
      *(float4*)&As[r * LDA + c8] =
          *(const float4*)&AO[(size_t)(row0 + r) * 1024 + kt + c8];
    }
    // B tile: 32x64 f32 -> f16 transposed to [n][k], b128 global loads
#pragma unroll
    for (int i = 0; i < 2; ++i) {
      int e = i * 256 + tid;
      int k = e >> 4, n4 = (e & 15) * 4;
      float4 f = *(const float4*)&Wout[(size_t)(kt + k) * 1024 + col0 + n4];
      Bs[(n4 + 0) * LDA + k] = (_Float16)f.x;
      Bs[(n4 + 1) * LDA + k] = (_Float16)f.y;
      Bs[(n4 + 2) * LDA + k] = (_Float16)f.z;
      Bs[(n4 + 3) * LDA + k] = (_Float16)f.w;
    }
    if (kt + TK < 1024) {
      __builtin_prefetch(&AO[(size_t)(row0 + (tid >> 1)) * 1024 + kt + TK + (tid & 1) * 16], 0, 1);
      __builtin_prefetch(&Wout[(size_t)(kt + TK + (tid >> 6)) * 1024 + col0 + (tid & 63)], 0, 1);
    }
    __syncthreads();
    v16h a0 = frag_a(&As[wr * LDA], LDA);
    v16h a1 = frag_a(&As[(wr + 16) * LDA], LDA);
    v16h b0 = frag_b(&Bs[wc * LDA], LDA);
    v16h b1 = frag_b(&Bs[(wc + 16) * LDA], LDA);
    acc00 = WMMA_F32_F16(a0, b0, acc00);
    acc01 = WMMA_F32_F16(a0, b1, acc01);
    acc10 = WMMA_F32_F16(a1, b0, acc10);
    acc11 = WMMA_F32_F16(a1, b1, acc11);
    __syncthreads();
  }
  const int lane = tid & 31;
  const int n = lane & 15, hf = lane >> 4;
#pragma unroll
  for (int ct = 0; ct < 2; ++ct) {
    int col = col0 + wc + ct * 16 + n;
    float bias = bout[col];
#pragma unroll
    for (int rt = 0; rt < 2; ++rt) {
      size_t obase = (size_t)(row0 + wr + rt * 16 + hf * 8) * 1024 + col;
#pragma unroll
      for (int r = 0; r < 8; ++r) {
        out[obase + (size_t)r * 1024] =
            ((ct == 0) ? (rt == 0 ? acc00[r] : acc10[r])
                       : (rt == 0 ? acc01[r] : acc11[r])) + bias;
      }
    }
  }
}

// ---------------- Launch ----------------
extern "C" void kernel_launch(void* const* d_in, const int* in_sizes, int n_in,
                              void* d_out, int out_size, void* d_ws, size_t ws_size,
                              hipStream_t stream) {
  (void)in_sizes; (void)n_in; (void)out_size; (void)ws_size;
  const float* x    = (const float*)d_in[0];
  const float* Wqkv = (const float*)d_in[1];
  const float* bqkv = (const float*)d_in[2];
  const float* Wout = (const float*)d_in[3];
  const float* bout = (const float*)d_in[4];
  float* out = (float*)d_out;

  // workspace: Q,K,V f16 [B,H,T,D] + attn-out f16 [B,T,C]  (4 x 8MB = 32MB)
  _Float16* ws = (_Float16*)d_ws;
  const size_t NE = (size_t)2 * 16 * 2048 * 64;
  _Float16* Qh = ws;
  _Float16* Kh = ws + NE;
  _Float16* Vh = ws + 2 * NE;
  _Float16* AO = ws + 3 * NE;

  dim3 blk(256);
  hipLaunchKernelGGL(qkv_gemm, dim3(48, 32), blk, 0, stream, x, Wqkv, bqkv, Qh, Kh, Vh);
  hipLaunchKernelGGL(attention, dim3(16, 32), blk, 0, stream, Qh, Kh, Vh, AO);
  hipLaunchKernelGGL(out_gemm, dim3(16, 32), blk, 0, stream, AO, Wout, bout, out);
}
